// ReLULSTM_29609504539459
// MI455X (gfx1250) — matrix operand order repaired and checked by reading, not compiled, not run on hardware
//
#include <hip/hip_runtime.h>

// Problem constants (match reference)
#define B_   64
#define T_   512
#define DIN  512
#define H_   1024
#define G4   4096   // 4*H
#define NL   2

typedef __attribute__((ext_vector_type(16))) __bf16 v16bf;
typedef __attribute__((ext_vector_type(8)))  float  v8f;

// ---------------------------------------------------------------------------
// Convert + transpose a weight matrix: W[K][4096] f32 (row-major, as in the
// reference: x @ W) -> Wt[4096][K] bf16 so that the WMMA B-fragment
// (per-lane contiguous K range of 16) is a contiguous 32-byte load.
// ---------------------------------------------------------------------------
__global__ void cvt_weight_T(const float* __restrict__ W,
                             __bf16* __restrict__ Wt, int K) {
    long idx = (long)blockIdx.x * blockDim.x + threadIdx.x;  // over output
    long total = (long)K * G4;
    if (idx >= total) return;
    int n = (int)(idx / K);   // gate column 0..4095
    int k = (int)(idx % K);
    Wt[idx] = (__bf16)W[(long)k * G4 + n];
}

// Plain f32 -> bf16 elementwise (for the input sequence x)
__global__ void cvt_bf16(const float* __restrict__ in,
                         __bf16* __restrict__ out, long n) {
    long idx = (long)blockIdx.x * blockDim.x + threadIdx.x;
    if (idx < n) out[idx] = (__bf16)in[idx];
}

// Combined bias: bias[g] = b_ih[g] + b_hh[g]
__global__ void add_bias(const float* __restrict__ a,
                         const float* __restrict__ b,
                         float* __restrict__ o) {
    int i = blockIdx.x * blockDim.x + threadIdx.x;
    if (i < G4) o[i] = a[i] + b[i];
}

// Fast activations on the v_exp_f32 / v_rcp_f32 TRANS pipes
__device__ __forceinline__ float fast_sigmoid(float x) {
    return __builtin_amdgcn_rcpf(1.0f + __expf(-x));
}
__device__ __forceinline__ float fast_tanh(float x) {
    // 1 - 2/(exp(2x)+1): exact limits at +/-inf, no divide sequence
    return 1.0f - 2.0f * __builtin_amdgcn_rcpf(__expf(2.0f * x) + 1.0f);
}

// ---------------------------------------------------------------------------
// One LSTM layer step:  gates = Xbf @ WihT^T + Hbf @ WhhT^T + bias
// then the ReLU-cell LSTM pointwise update.
//
// grid = (16, 4): blockIdx.x = h-column tile of 64, blockIdx.y = batch tile
// of 16. block = 128 threads = 4 waves; wave g computes gate g's 16x64 tile
// as four v_wmma_f32_16x16x32_bf16 accumulators, gates meet in LDS.
// ---------------------------------------------------------------------------
__launch_bounds__(128)
__global__ void lstm_step(
    const __bf16* __restrict__ Xbf,   int xstride, int Kx,  // [B_, Kx], row stride xstride
    const __bf16* __restrict__ Hbf,                          // [B_, H_] prev hidden (bf16)
    const __bf16* __restrict__ WihT,                         // [G4][Kx] bf16 transposed
    const __bf16* __restrict__ WhhT,                         // [G4][H_] bf16 transposed
    const float*  __restrict__ bias,                         // [G4] b_ih + b_hh
    float*        __restrict__ C,                            // [B_][H_] cell state (in/out)
    float*        __restrict__ Hout_f,                       // [B_][H_] fp32 hidden (write-only)
    __bf16*       __restrict__ Hout_bf,                      // [B_][H_] bf16 hidden (next-step feed)
    float*        __restrict__ seq_out)                      // &output[0][t][0] (row stride T_*H_) or null
{
    const int lane = threadIdx.x & 31;
    const int wave = threadIdx.x >> 5;        // gate id: 0=i 1=f 2=c 3=o
    const int half = lane >> 4;
    const int l16  = lane & 15;

    const int row0  = blockIdx.y * 16;        // batch rows
    const int hcol0 = blockIdx.x * 64;        // hidden cols
    const int gcol0 = wave * H_ + hcol0;      // gate column base in [0,4096)

    v8f acc[4] = {v8f{}, v8f{}, v8f{}, v8f{}};

    // Two GEMM phases: input projection (Kx) and recurrent projection (H_)
    #pragma unroll 1
    for (int phase = 0; phase < 2; ++phase) {
        const __bf16* Arow = (phase == 0)
            ? (Xbf + (long)(row0 + l16) * xstride)
            : (Hbf + (long)(row0 + l16) * H_);
        const __bf16* Wt = (phase == 0) ? WihT : WhhT;
        const int K = (phase == 0) ? Kx : H_;

        for (int k0 = 0; k0 < K; k0 += 32) {
            // ---- A fragment: 16x32 bf16 (contiguous K: lane-half h holds
            //      k = 16*h .. 16*h+15, matching B) ----
            v16bf a;
            #pragma unroll
            for (int v = 0; v < 8; ++v) {
                int kk = k0 + (v * 2) + (half * 16);
                a[2 * v]     = Arow[kk];
                a[2 * v + 1] = Arow[kk + 1];
            }
            // ---- 4 B fragments (32x16 each) + 4 WMMAs sharing A ----
            #pragma unroll
            for (int j = 0; j < 4; ++j) {
                const __bf16* wp =
                    Wt + (long)(gcol0 + 16 * j + l16) * K + k0 + 16 * half;
                v16bf b;
                #pragma unroll
                for (int e = 0; e < 16; ++e) b[e] = wp[e];
                acc[j] = __builtin_amdgcn_wmma_f32_16x16x32_bf16(
                    false, a, false, b, (short)0, acc[j], false, false);
            }
        }
    }

    // ---- add bias, apply per-gate activation, stage in LDS.
    //      Gate type is wave-uniform: hoist the branch OUTSIDE the unrolled
    //      store loop so it compiles to one s_cbranch + two straight bodies.
    __shared__ float lds_g[4][16][64];       // 16 KB
    if (__builtin_amdgcn_readfirstlane(wave) == 2) {
        #pragma unroll
        for (int j = 0; j < 4; ++j) {
            #pragma unroll
            for (int v = 0; v < 8; ++v) {
                int r = v + half * 8;        // C/D layout: M = v + 8*half
                int n = 16 * j + l16;        // N = lane%16 within subtile
                lds_g[wave][r][n] = fmaxf(acc[j][v] + bias[gcol0 + n], 0.0f);
            }
        }
    } else {
        #pragma unroll
        for (int j = 0; j < 4; ++j) {
            #pragma unroll
            for (int v = 0; v < 8; ++v) {
                int r = v + half * 8;
                int n = 16 * j + l16;
                lds_g[wave][r][n] = fast_sigmoid(acc[j][v] + bias[gcol0 + n]);
            }
        }
    }
    __syncthreads();

    // ---- pointwise LSTM update: 16x64 elements, 8 per thread ----
    for (int e = threadIdx.x; e < 16 * 64; e += 128) {
        int r = e >> 6, n = e & 63;
        int b = row0 + r;
        int h = hcol0 + n;
        long idx = (long)b * H_ + h;
        float ig = lds_g[0][r][n];
        float fg = lds_g[1][r][n];
        float cg = lds_g[2][r][n];
        float og = lds_g[3][r][n];
        float c_new = fg * C[idx] + ig * cg;
        float h_new = og * fast_tanh(c_new);
        C[idx]       = c_new;
        Hout_f[idx]  = h_new;
        Hout_bf[idx] = (__bf16)h_new;
        if (seq_out) seq_out[(long)b * T_ * H_ + h] = h_new;
    }
}

// ---------------------------------------------------------------------------
extern "C" void kernel_launch(void* const* d_in, const int* in_sizes, int n_in,
                              void* d_out, int out_size, void* d_ws, size_t ws_size,
                              hipStream_t stream) {
    (void)in_sizes; (void)n_in; (void)out_size; (void)ws_size;

    const float* x     = (const float*)d_in[0];
    const float* W_ih0 = (const float*)d_in[1];
    const float* b_ih0 = (const float*)d_in[2];
    const float* W_hh0 = (const float*)d_in[3];
    const float* b_hh0 = (const float*)d_in[4];
    const float* W_ih1 = (const float*)d_in[5];
    const float* b_ih1 = (const float*)d_in[6];
    const float* W_hh1 = (const float*)d_in[7];
    const float* b_hh1 = (const float*)d_in[8];
    float* out = (float*)d_out;

    // ---- workspace carve-out (~65 MB) ----
    char* ws = (char*)d_ws;
    size_t off = 0;
    auto carve = [&](size_t bytes) {
        void* p = ws + off;
        off += (bytes + 255) & ~(size_t)255;
        return p;
    };
    __bf16* WihT0 = (__bf16*)carve((size_t)DIN * G4 * 2);
    __bf16* WhhT0 = (__bf16*)carve((size_t)H_  * G4 * 2);
    __bf16* WihT1 = (__bf16*)carve((size_t)H_  * G4 * 2);
    __bf16* WhhT1 = (__bf16*)carve((size_t)H_  * G4 * 2);
    __bf16* Xbf   = (__bf16*)carve((size_t)B_ * T_ * DIN * 2);
    __bf16* h0bf[2] = {(__bf16*)carve((size_t)B_ * H_ * 2),
                       (__bf16*)carve((size_t)B_ * H_ * 2)};
    __bf16* h1bf[2] = {(__bf16*)carve((size_t)B_ * H_ * 2),
                       (__bf16*)carve((size_t)B_ * H_ * 2)};
    float* h0f   = (float*)carve((size_t)B_ * H_ * 4);
    float* h1f   = (float*)carve((size_t)B_ * H_ * 4);
    float* c0    = (float*)carve((size_t)B_ * H_ * 4);
    float* c1    = (float*)carve((size_t)B_ * H_ * 4);
    float* bias0 = (float*)carve((size_t)G4 * 4);
    float* bias1 = (float*)carve((size_t)G4 * 4);

    // ---- one-time (per launch) conversions ----
    {
        long n0 = (long)DIN * G4, n1 = (long)H_ * G4;
        cvt_weight_T<<<(unsigned)((n0 + 255) / 256), 256, 0, stream>>>(W_ih0, WihT0, DIN);
        cvt_weight_T<<<(unsigned)((n1 + 255) / 256), 256, 0, stream>>>(W_hh0, WhhT0, H_);
        cvt_weight_T<<<(unsigned)((n1 + 255) / 256), 256, 0, stream>>>(W_ih1, WihT1, H_);
        cvt_weight_T<<<(unsigned)((n1 + 255) / 256), 256, 0, stream>>>(W_hh1, WhhT1, H_);
        long nx = (long)B_ * T_ * DIN;
        cvt_bf16<<<(unsigned)((nx + 255) / 256), 256, 0, stream>>>(x, Xbf, nx);
        add_bias<<<G4 / 256, 256, 0, stream>>>(b_ih0, b_hh0, bias0);
        add_bias<<<G4 / 256, 256, 0, stream>>>(b_ih1, b_hh1, bias1);
    }

    // ---- zero initial states (graph-capture safe) ----
    hipMemsetAsync(h0bf[0], 0, (size_t)B_ * H_ * 2, stream);
    hipMemsetAsync(h1bf[0], 0, (size_t)B_ * H_ * 2, stream);
    hipMemsetAsync(c0,      0, (size_t)B_ * H_ * 4, stream);
    hipMemsetAsync(c1,      0, (size_t)B_ * H_ * 4, stream);

    // ---- sequential time loop: two fused GEMM+pointwise kernels per step ----
    dim3 grid(16, 4, 1);
    for (int t = 0; t < T_; ++t) {
        const __bf16* h0r = h0bf[t & 1];
        __bf16*       h0w = h0bf[(t + 1) & 1];
        const __bf16* h1r = h1bf[t & 1];
        __bf16*       h1w = h1bf[(t + 1) & 1];

        // layer 0: input x[:, t, :] (row stride T_*DIN), K = DIN
        lstm_step<<<grid, 128, 0, stream>>>(
            Xbf + (size_t)t * DIN, T_ * DIN, DIN,
            h0r, WihT0, WhhT0, bias0,
            c0, h0f, h0w, nullptr);

        // layer 1: input = layer-0 hidden just written, K = H_
        lstm_step<<<grid, 128, 0, stream>>>(
            h0w, H_, H_,
            h1r, WihT1, WhhT1, bias1,
            c1, h1f, h1w, out + (size_t)t * H_);
    }

    // ---- final states: h_n = [h0; h1], c_n = [c0; c1] after the output ----
    float* hn = out + (size_t)B_ * T_ * H_;
    float* cn = hn + (size_t)NL * B_ * H_;
    size_t st = (size_t)B_ * H_ * sizeof(float);
    hipMemcpyAsync(hn,           h0f, st, hipMemcpyDeviceToDevice, stream);
    hipMemcpyAsync(hn + B_ * H_, h1f, st, hipMemcpyDeviceToDevice, stream);
    hipMemcpyAsync(cn,           c0,  st, hipMemcpyDeviceToDevice, stream);
    hipMemcpyAsync(cn + B_ * H_, c1,  st, hipMemcpyDeviceToDevice, stream);
}
